// CRBM_29807073034602
// MI455X (gfx1250) — matrix operand order repaired and checked by reading, not compiled, not run on hardware
//
#include <hip/hip_runtime.h>
#include <hip/hip_bf16.h>

// ---------------------------------------------------------------------------
// CDNA5 / gfx1250 CRBM CD-1 step.
//  - All matrix math via v_wmma_f32_16x16x32_f16 (wave32).
//  - All large intermediates stored f16 (HBM roofline: ~1 GB @ 23.3 TB/s).
//  - LDS staging through the Tensor Data Mover where available.
//  - All WMMA fragments load as contiguous ds_load_b128/b32.
// ---------------------------------------------------------------------------

typedef __attribute__((ext_vector_type(16))) _Float16 v16h;
typedef __attribute__((ext_vector_type(8)))  float    v8f;
typedef unsigned u32x4 __attribute__((ext_vector_type(4)));

union FragU {
  v16h  v;
  u32x4 q[2];
  unsigned d[8];
};

__device__ __forceinline__ v8f wmma_f16(v16h a, v16h b, v8f c) {
  return __builtin_amdgcn_wmma_f32_16x16x32_f16(false, a, false, b, (short)0, c,
                                                false, false);
}

// ISA 7.12.2: 16-bit A 16x32 (MxK), wave32: lane L: m=L&15, half=L>>4.
// v16h element e<8  <-> K = half*8 + e        (contiguous)
// v16h element e>=8 <-> K = 16 + half*8 + e-8 (contiguous)
__device__ __forceinline__ int kidxA(int lane, int j, int t) {
  int half = lane >> 4;
  return (j < 4) ? (half * 8 + 2 * j + t) : (16 + half * 8 + 2 * (j - 4) + t);
}
// B 32x16 (KxN), this code's contiguous k order: lane L: n=L&15, koff=(L>>4)*16; element e <-> K = koff + e.
__device__ __forceinline__ int kidxB(int lane, int j, int t) {
  return ((lane >> 4) << 4) + 2 * j + t;
}

__device__ __forceinline__ unsigned pack2(float lo, float hi) {
  union { _Float16 h; unsigned short s; } a, b;
  a.h = (_Float16)lo; b.h = (_Float16)hi;
  return (unsigned)a.s | ((unsigned)b.s << 16);
}

__device__ __forceinline__ float sigmoidf_(float z) {
  return 1.0f / (1.0f + __expf(-z));
}

// ---------------------------------------------------------------------------
// Tensor Data Mover wrapper (CDNA5 ISA D# layout).
// ---------------------------------------------------------------------------
#if defined(__has_builtin)
#if __has_builtin(__builtin_amdgcn_tensor_load_to_lds) && \
    __has_builtin(__builtin_amdgcn_s_wait_tensorcnt)
#define HAVE_TDM 1
#endif
#endif
#ifndef HAVE_TDM
#define HAVE_TDM 0
#endif

__device__ __forceinline__ unsigned lds_addr_of(const void* p) {
  // Generic LDS pointer: low 32 bits are the LDS byte offset (ISA aperture map).
  return (unsigned)(unsigned long long)(uintptr_t)p;
}

#if HAVE_TDM
typedef unsigned tdm_u4 __attribute__((ext_vector_type(4)));
typedef int      tdm_i4 __attribute__((ext_vector_type(4)));
typedef int      tdm_i8 __attribute__((ext_vector_type(8)));

// 2-byte elements; up-to-3D tile load (tile2==0 -> 2D).  Addressing:
//   global = base + 2*(x + y*stride0 + z*stride1); OOB reads return zero.
__device__ __forceinline__ void tdm_load(
    unsigned lds, const void* gptr, unsigned dim0, unsigned dim1, unsigned dim2,
    unsigned long long stride0, unsigned long long stride1,
    unsigned tile0, unsigned tile1, unsigned tile2, unsigned padbits) {
  unsigned long long ga = (unsigned long long)gptr;
  tdm_u4 g0;
  g0[0] = 1u;                                   // count=1 (valid user D#)
  g0[1] = lds;                                  // lds_addr (bytes)
  g0[2] = (unsigned)ga;                         // global_addr[31:0]
  g0[3] = (unsigned)((ga >> 32) & 0x01ffffffu) | (2u << 30);  // [56:32] | type=2
  tdm_i8 g1;
  g1[0] = (int)((1u << 16) | padbits);          // data_size=1 (2B) | pad ctl
  g1[1] = (int)((dim0 & 0xffffu) << 16);
  g1[2] = (int)(((dim0 >> 16) & 0xffffu) | ((dim1 & 0xffffu) << 16));
  g1[3] = (int)(((dim1 >> 16) & 0xffffu) | ((tile0 & 0xffffu) << 16));
  g1[4] = (int)((tile1 & 0xffffu) | ((tile2 & 0xffffu) << 16));
  g1[5] = (int)(unsigned)(stride0 & 0xffffffffu);
  g1[6] = (int)(((unsigned)(stride0 >> 32) & 0xffffu) |
                (((unsigned)stride1 & 0xffffu) << 16));
  g1[7] = (int)(unsigned)((stride1 >> 16) & 0xffffffffu);
  tdm_i4 g2; g2[0] = (int)dim2; g2[1] = 0; g2[2] = 0; g2[3] = 0;
  tdm_i4 g3 = (tdm_i4)0;
#if __clang_major__ >= 23
  tdm_i8 g4 = (tdm_i8)0;
  __builtin_amdgcn_tensor_load_to_lds(g0, g1, g2, g3, g4, 0);
#else
  __builtin_amdgcn_tensor_load_to_lds(g0, g1, g2, g3, 0);
#endif
}
#endif  // HAVE_TDM

// ---------------------------------------------------------------------------
// Pack W into WMMA fragment layouts (f16).
// ---------------------------------------------------------------------------
__global__ __launch_bounds__(256) void prep_pack_kernel(
    const float* __restrict__ W, unsigned* __restrict__ Apack,
    unsigned* __restrict__ Bpack) {
  int tid = blockIdx.x * 256 + threadIdx.x;
  if (tid < 1024) {
    int lane = tid & 31, rest = tid >> 5;
    int otile = rest >> 3, ks = rest & 7;
    int o = otile * 16 + (lane & 15);
#pragma unroll
    for (int j = 0; j < 8; ++j) {
      float lo = 0.f, hi = 0.f;
      int k0 = ks * 32 + kidxA(lane, j, 0);
      int k1 = ks * 32 + kidxA(lane, j, 1);
      if (k0 < 243) { int c = k0 / 81, r = k0 % 81; lo = W[((o * 3 + c) * 9 + r / 9) * 9 + r % 9]; }
      if (k1 < 243) { int c = k1 / 81, r = k1 % 81; hi = W[((o * 3 + c) * 9 + r / 9) * 9 + r % 9]; }
      Apack[tid * 8 + j] = pack2(lo, hi);
    }
  }
  if (tid < 5184) {
    int lane = tid & 31, ks = tid >> 5;
    int n = lane & 15;
#pragma unroll
    for (int j = 0; j < 8; ++j) {
      float lo = 0.f, hi = 0.f;
      if (n < 3) {
        int kk0 = ks * 32 + kidxB(lane, j, 0);
        int kk1 = ks * 32 + kidxB(lane, j, 1);
        { int p = kk0 >> 6, o = kk0 & 63, i = p / 9, jx = p % 9;
          lo = W[((o * 3 + n) * 9 + (8 - i)) * 9 + (8 - jx)]; }
        { int p = kk1 >> 6, o = kk1 & 63, i = p / 9, jx = p % 9;
          hi = W[((o * 3 + n) * 9 + (8 - i)) * 9 + (8 - jx)]; }
      }
      Bpack[tid * 8 + j] = pack2(lo, hi);
    }
  }
}

__global__ __launch_bounds__(256) void convert_v_kernel(
    const float* __restrict__ v, _Float16* __restrict__ vh) {
  const int N = 16 * 3 * 65536;
  for (int i = blockIdx.x * 256 + threadIdx.x; i < N; i += gridDim.x * 256)
    vh[i] = (_Float16)v[i];
}

// ---------------------------------------------------------------------------
// Forward conv (implicit GEMM, M=o, K=243->256, N=32 x-positions of one row).
// im2col is materialized once per workgroup into K-major patchT, so every
// B fragment is 2 aligned ds_load_b128 (no per-element gathers in K loop).
// ---------------------------------------------------------------------------
__global__ __launch_bounds__(256) void conv_fwd_wmma(
    const _Float16* __restrict__ vin, const unsigned* __restrict__ Apack,
    const float* __restrict__ hbias, const float* __restrict__ u,
    _Float16* __restrict__ hprob, _Float16* __restrict__ hsamp) {
  __shared__ __align__(16) _Float16 vrow[3][9][40];
  __shared__ __align__(16) _Float16 patchT[32][256];  // [position][K]
  const int tid = threadIdx.x;
  const int lane = tid & 31, w = tid >> 5;
  const int hf = lane >> 4;
  const int x0 = blockIdx.x * 32;
  const int y = blockIdx.y;
  const int b = blockIdx.z;

#if HAVE_TDM
  if (w == 0) {
    const _Float16* g = vin + ((size_t)(b * 3) * 65536 + (size_t)y * 256 + x0);
    tdm_load(lds_addr_of(vrow), g, /*dims*/ 256 - x0, 9, 3,
             /*strides*/ 256, 65536, /*tile*/ 40, 9, 3, 0u);
    __builtin_amdgcn_s_wait_tensorcnt(0);
  }
#else
  for (int e = tid; e < 3 * 9 * 40; e += 256) {
    int c = e / 360; int rem = e - c * 360; int ky = rem / 40; int xx = rem - ky * 40;
    int gx = x0 + xx;
    vrow[c][ky][xx] = (gx < 256)
        ? vin[((size_t)(b * 3 + c) * 256 + (y + ky)) * 256 + gx]
        : (_Float16)0.f;
  }
#endif
  __syncthreads();

  // Each thread owns one K column: gather decomposition computed once.
  {
    const int kk = tid;  // 0..255
    if (kk < 243) {
      int c = kk / 81; int r = kk - c * 81; int ky = r / 9; int kx = r - ky * 9;
      const _Float16* src = &vrow[c][ky][kx];
#pragma unroll
      for (int x = 0; x < 32; ++x) patchT[x][kk] = src[x];
    } else {
#pragma unroll
      for (int x = 0; x < 32; ++x) patchT[x][kk] = (_Float16)0.f;
    }
  }
  __syncthreads();

  const int otile = w & 3, ntile = w >> 2;
  const int n = lane & 15;
  const int nb = ntile * 16 + n;
  const _Float16* pB = &patchT[nb][0];   // 512B rows, 16B-aligned chunks
  v8f acc = {};
#pragma unroll
  for (int ks = 0; ks < 8; ++ks) {
    v16h a = *reinterpret_cast<const v16h*>(
        Apack + (((otile * 8 + ks) * 32 + lane) << 3));
    FragU fb;
    fb.q[0] = *reinterpret_cast<const u32x4*>(pB + ks * 32 + hf * 8);
    fb.q[1] = *reinterpret_cast<const u32x4*>(pB + ks * 32 + 16 + hf * 8);
    acc = wmma_f16(a, fb.v, acc);
  }

  const int xcol = x0 + nb;
  if (xcol < 248) {
#pragma unroll
    for (int r = 0; r < 8; ++r) {
      int o = otile * 16 + (hf * 8 + r);   // D: m = r + 8*(lane>>4)
      size_t idx = ((size_t)(b * 64 + o) * 248 + y) * 248 + xcol;
      float p = sigmoidf_(acc[r] + hbias[o]);
      hprob[idx] = (_Float16)p;
      if (hsamp) hsamp[idx] = (_Float16)((u[idx] < p) ? 1.0f : 0.0f);
    }
  }
}

// ---------------------------------------------------------------------------
// Conv-transpose (implicit GEMM): D[x, c] = sum_K hpad[x,K] * Wflip[K,c].
// Per (i,jx) step: stage all 64 o-planes (two WMMAs per stage, 81 stages).
// Staging loads are 16B u32x4 chunks (8x fewer VMEM instructions); hbufT is
// position-major so each A fragment is 2 aligned ds_load_b128.
// ---------------------------------------------------------------------------
__global__ __launch_bounds__(256) void conv_bwd_wmma(
    const _Float16* __restrict__ hsamp, const unsigned* __restrict__ Bpack,
    const float* __restrict__ vbias, const float* __restrict__ uv,
    _Float16* __restrict__ vprob, _Float16* __restrict__ vsamp) {
  __shared__ __align__(16) _Float16 hbufT[128][64];  // [position][o]
  const int tid = threadIdx.x;
  const int lane = tid & 31, w = tid >> 5;
  const int hf = lane >> 4;
  const int x0 = blockIdx.x * 128;
  const int yy = blockIdx.y;
  const int b = blockIdx.z;

  const _Float16* abase = &hbufT[w * 16 + (lane & 15)][0];  // 128B rows
  v8f acc = {};
  for (int ps = 0; ps < 81; ++ps) {
    const int i = ps / 9, jx = ps - i * 9;
    const int row = yy + i - 8;
    const bool rok = (unsigned)row < 248u;
    const int colbase = x0 + jx - 8;
    const int colA = colbase & ~7;       // 16B-aligned slab base
    const int s = colbase - colA;        // 0..7
    __syncthreads();
    for (int e2 = tid; e2 < 64 * 17; e2 += 256) {
      int ol = e2 / 17, ch = e2 - ol * 17;   // ch consecutive -> coalesced
      union { u32x4 q; _Float16 h[8]; } U;
      U.q = (u32x4)0;
      if (rok) {
        const _Float16* src =
            hsamp + ((size_t)(b * 64 + ol) * 248 + row) * 248;
        U.q = *reinterpret_cast<const u32x4*>(src + colA + ch * 8);
      }
#pragma unroll
      for (int e = 0; e < 8; ++e) {
        int m = ch * 8 + e - s;
        if ((unsigned)m < 128u) {
          int col = colbase + m;
          hbufT[m][ol] =
              (rok && (unsigned)col < 248u) ? U.h[e] : (_Float16)0.f;
        }
      }
    }
    __syncthreads();
#pragma unroll
    for (int half = 0; half < 2; ++half) {   // o 0..31, then o 32..63
      FragU fa;
      fa.q[0] = *reinterpret_cast<const u32x4*>(abase + half * 32 + hf * 16);
      fa.q[1] = *reinterpret_cast<const u32x4*>(abase + half * 32 + hf * 16 + 8);
      v16h bm = *reinterpret_cast<const v16h*>(
          Bpack + (((ps * 2 + half) * 32 + lane) << 3));
      acc = wmma_f16(fa.v, bm, acc);
    }
  }

  const int c = lane & 15;
  if (c < 3) {
#pragma unroll
    for (int r = 0; r < 8; ++r) {
      int xp = x0 + w * 16 + (hf * 8 + r);
      size_t idx = ((size_t)(b * 3 + c) * 256 + yy) * 256 + xp;
      float pv = sigmoidf_(acc[r] + vbias[c]);
      vprob[idx] = (_Float16)pv;
      vsamp[idx] = (_Float16)((uv[idx] < pv) ? 1.0f : 0.0f);
    }
  }
}

// ---------------------------------------------------------------------------
// Correlation: accum[j,q] += sum_{b,y,x} hh[b,j,y,x] * vv[b,c(q),y+dy,x+dx].
// hrow rows = 496B (16B aligned chunks) -> A fragment = 2 ds_load_b128.
// vrow + one-element-shifted vrowS give 4B-aligned contiguous B fragments
// for both dx parities -> B fragment = 8 ds_load_b32.
// ---------------------------------------------------------------------------
__global__ __launch_bounds__(256) void corr_wmma(
    const _Float16* __restrict__ vv, const _Float16* __restrict__ hh,
    float* __restrict__ accum) {
  __shared__ __align__(16) _Float16 hrow[64][248];
  __shared__ __align__(16) _Float16 vrow[3][9][264];   // vrow [x] = vv[x]
  __shared__ __align__(16) _Float16 vrowS[3][9][264];  // vrowS[x] = vv[x+1]
  const int tid = threadIdx.x;
  const int lane = tid & 31, w = tid >> 5;
  const int hf = lane >> 4;
  const int ybase = blockIdx.x * 8;
  const int b = blockIdx.y;
  const int m = lane & 15;

  // zero the pad columns once (TDM pad skips them; they must read as 0)
  for (int e = tid; e < 27 * 8; e += 256) {
    int r = e >> 3; int c = r / 9; int dy = r - c * 9;
    vrow[c][dy][256 + (e & 7)]  = (_Float16)0.f;
    vrowS[c][dy][256 + (e & 7)] = (_Float16)0.f;
  }

  // hoist per-lane q -> (c,dy,dx); pick parity-aligned B base pointer
  bool okq[2]; const _Float16* vb[2]; int qn[2];
#pragma unroll
  for (int qq = 0; qq < 2; ++qq) {
    int q = (w * 2 + qq) * 16 + m;
    qn[qq] = q;
    okq[qq] = q < 243;
    int c = 0, dy = 0, dx = 0;
    if (okq[qq]) { c = q / 81; int rr = q - c * 81; dy = rr / 9; dx = rr - dy * 9; }
    vb[qq] = (dx & 1) ? &vrowS[c][dy][dx - 1] : &vrow[c][dy][dx];
  }
  const _Float16* abase[4];
#pragma unroll
  for (int jt = 0; jt < 4; ++jt) abase[jt] = &hrow[jt * 16 + m][0];

  v8f acc[4][2] = {};
  for (int ry = 0; ry < 8; ++ry) {
    const int y = ybase + ry;
    __syncthreads();
#if HAVE_TDM
    if (w == 0) {
      const _Float16* gh = hh + ((size_t)(b * 64) * 61504 + (size_t)y * 248);
      tdm_load(lds_addr_of(hrow), gh, 248, 64, 1, 61504, 0, 248, 64, 0, 0u);
      const unsigned pad = (1u << 20) | (6u << 22) | (3u << 25);  // 128DW + 4DW
      for (int c = 0; c < 3; ++c) {
        const _Float16* gv = vv + ((size_t)(b * 3 + c) * 65536 + (size_t)y * 256);
        tdm_load(lds_addr_of(vrow[c]),  gv,     256, 9, 1, 256, 0, 256, 9, 0, pad);
        tdm_load(lds_addr_of(vrowS[c]), gv + 1, 255, 9, 1, 256, 0, 256, 9, 0, pad);
      }
      __builtin_amdgcn_s_wait_tensorcnt(0);
    }
#else
    for (int e = tid; e < 64 * 248; e += 256) {
      int j = e / 248; int x = e - j * 248;
      hrow[j][x] = hh[((size_t)(b * 64 + j) * 248 + y) * 248 + x];
    }
    for (int e = tid; e < 27 * 256; e += 256) {
      int r = e >> 8; int c = r / 9; int dy = r - c * 9; int x = e & 255;
      const _Float16* src = vv + ((size_t)(b * 3 + c) * 65536 + (size_t)(y + dy) * 256);
      vrow[c][dy][x]  = src[x];
      vrowS[c][dy][x] = (x < 255) ? src[x + 1] : (_Float16)0.f;
    }
#endif
    __syncthreads();

#pragma unroll
    for (int ks = 0; ks < 8; ++ks) {
      FragU fa[4];
#pragma unroll
      for (int jt = 0; jt < 4; ++jt) {
        fa[jt].q[0] = *reinterpret_cast<const u32x4*>(abase[jt] + ks * 32 + hf * 16);
        fa[jt].q[1] = *reinterpret_cast<const u32x4*>(abase[jt] + ks * 32 + hf * 16 + 8);
        if (ks == 7) {
          // chunk1 of half-1 lanes covers x=248..255 -> zero (K tail mask)
          if (hf) fa[jt].q[1] = (u32x4)0;
        }
      }
#pragma unroll
      for (int qq = 0; qq < 2; ++qq) {
        FragU fb;
        const _Float16* p = vb[qq] + ks * 32 + hf * 16;  // even offset: 4B aligned
#pragma unroll
        for (int j = 0; j < 8; ++j)
          fb.d[j] = *reinterpret_cast<const unsigned*>(p + 2 * j);
        if (!okq[qq]) { fb.q[0] = (u32x4)0; fb.q[1] = (u32x4)0; }
#pragma unroll
        for (int jt = 0; jt < 4; ++jt)
          acc[jt][qq] = wmma_f16(fa[jt].v, fb.v, acc[jt][qq]);
      }
    }
  }

#pragma unroll
  for (int jt = 0; jt < 4; ++jt)
#pragma unroll
    for (int qq = 0; qq < 2; ++qq)
      if (okq[qq]) {
#pragma unroll
        for (int r = 0; r < 8; ++r) {
          int jch = jt * 16 + (hf * 8 + r);
          atomicAdd(&accum[jch * 243 + qn[qq]], acc[jt][qq][r]);
        }
      }
}

// ---------------------------------------------------------------------------
// Reductions + finalize.
// ---------------------------------------------------------------------------
__global__ __launch_bounds__(256) void zero_accum_kernel(
    float* __restrict__ accP, float* __restrict__ accN, float* __restrict__ misc) {
  int i = blockIdx.x * 256 + threadIdx.x;
  if (i < 15552) { accP[i] = 0.f; accN[i] = 0.f; }
  if (i < 68) misc[i] = 0.f;
}

__global__ __launch_bounds__(256) void reduce_v_kernel(
    const float* __restrict__ v, const _Float16* __restrict__ vs,
    float* __restrict__ misc) {
  float s0 = 0.f, s1 = 0.f, s2 = 0.f, sl = 0.f;
  const int N = 16 * 3 * 65536;
  for (int i = blockIdx.x * 256 + threadIdx.x; i < N; i += gridDim.x * 256) {
    int c = (i >> 16) % 3;
    float d = v[i] - (float)vs[i];
    if (c == 0) s0 += d; else if (c == 1) s1 += d; else s2 += d;
    sl += d * d;
  }
  __shared__ float red[256];
  float vals[4] = {s0, s1, s2, sl};
  for (int k = 0; k < 4; ++k) {
    red[threadIdx.x] = vals[k];
    __syncthreads();
    for (int off = 128; off > 0; off >>= 1) {
      if ((int)threadIdx.x < off) red[threadIdx.x] += red[threadIdx.x + off];
      __syncthreads();
    }
    if (threadIdx.x == 0) atomicAdd(&misc[k], red[0]);
    __syncthreads();
  }
}

__global__ __launch_bounds__(256) void reduce_h_kernel(
    const _Float16* __restrict__ h0, const _Float16* __restrict__ h1,
    float* __restrict__ misc) {
  const int plane = blockIdx.x;  // b*64 + o
  const int o = plane & 63;
  const size_t base = (size_t)plane * 61504;
  float s = 0.f;
  for (int i = threadIdx.x; i < 61504; i += 256) {
    __builtin_prefetch(&h0[base + i + 4096], 0, 1);  // global_prefetch_b8
    s += (float)h0[base + i] - (float)h1[base + i];
  }
  __shared__ float red[256];
  red[threadIdx.x] = s;
  __syncthreads();
  for (int off = 128; off > 0; off >>= 1) {
    if ((int)threadIdx.x < off) red[threadIdx.x] += red[threadIdx.x + off];
    __syncthreads();
  }
  if (threadIdx.x == 0) atomicAdd(&misc[4 + o], red[0]);
}

__global__ __launch_bounds__(256) void finalize_kernel(
    const float* __restrict__ W, const float* __restrict__ vb,
    const float* __restrict__ hb, const float* __restrict__ accP,
    const float* __restrict__ accN, const float* __restrict__ misc,
    float* __restrict__ out) {
  int idx = blockIdx.x * 256 + threadIdx.x;
  if (idx < 15552) {
    int o = idx / 243; int r = idx % 243; int p = r % 81;
    float s = 0.f;
#pragma unroll
    for (int c = 0; c < 3; ++c)
      s += accP[o * 243 + c * 81 + p] - accN[o * 243 + c * 81 + p];
    out[idx] = W[idx] + 0.01f * s / 16.0f;   // channel-broadcast per reference
  } else if (idx < 15555) {
    int c = idx - 15552;
    out[idx] = vb[c] + 0.01f * misc[c] / (16.0f * 256.0f * 256.0f);
  } else if (idx < 15619) {
    int o = idx - 15555;
    out[idx] = hb[o] + 0.01f * misc[4 + o] / (16.0f * 248.0f * 248.0f);
  } else if (idx == 15619) {
    out[idx] = misc[3] / (16.0f * 3.0f * 256.0f * 256.0f);
  }
}

// ---------------------------------------------------------------------------
extern "C" void kernel_launch(void* const* d_in, const int* in_sizes, int n_in,
                              void* d_out, int out_size, void* d_ws, size_t ws_size,
                              hipStream_t stream) {
  (void)in_sizes; (void)n_in; (void)out_size; (void)ws_size;
  const float* v     = (const float*)d_in[0];
  const float* W     = (const float*)d_in[1];
  const float* vbias = (const float*)d_in[2];
  const float* hbias = (const float*)d_in[3];
  const float* u_h0  = (const float*)d_in[4];
  const float* u_v1  = (const float*)d_in[5];
  // d_in[6] (u_h1) unused by the reference math.
  float* out = (float*)d_out;

  char* ws = (char*)d_ws;
  size_t off = 0;
  auto alloc = [&](size_t bytes) {
    size_t o = off;
    off = (off + bytes + 255) & ~(size_t)255;
    return o;
  };
  const size_t HPLANE = (size_t)16 * 64 * 248 * 248;  // 62,980,096
  const size_t VPLANE = (size_t)16 * 3 * 256 * 256;   //  3,145,728
  _Float16* hprob0 = (_Float16*)(ws + alloc(HPLANE * 2));
  _Float16* hprob1 = (_Float16*)(ws + alloc(HPLANE * 2));
  _Float16* hsamp0 = (_Float16*)(ws + alloc(HPLANE * 2));
  _Float16* vhalf  = (_Float16*)(ws + alloc(VPLANE * 2));
  _Float16* vprob1 = (_Float16*)(ws + alloc(VPLANE * 2));
  _Float16* vsamp1 = (_Float16*)(ws + alloc(VPLANE * 2));
  unsigned* Apack  = (unsigned*)(ws + alloc(8192 * 4));
  unsigned* Bpack  = (unsigned*)(ws + alloc(41472 * 4));
  float*    accP   = (float*)(ws + alloc(15552 * 4));
  float*    accN   = (float*)(ws + alloc(15552 * 4));
  float*    misc   = (float*)(ws + alloc(68 * 4));

  zero_accum_kernel<<<(15552 + 255) / 256, 256, 0, stream>>>(accP, accN, misc);
  prep_pack_kernel<<<(5184 + 255) / 256, 256, 0, stream>>>(W, Apack, Bpack);
  convert_v_kernel<<<1024, 256, 0, stream>>>(v, vhalf);
  // h_prob0 = sigmoid(conv(v,W)+hb); h_samp0 = (u_h0 < h_prob0)
  conv_fwd_wmma<<<dim3(8, 248, 16), 256, 0, stream>>>(vhalf, Apack, hbias, u_h0,
                                                      hprob0, hsamp0);
  // v_prob1 = sigmoid(convT(h_samp0,W)+vb); v_samp1 = (u_v1 < v_prob1)
  conv_bwd_wmma<<<dim3(2, 256, 16), 256, 0, stream>>>(hsamp0, Bpack, vbias, u_v1,
                                                      vprob1, vsamp1);
  // h_prob1 = sigmoid(conv(v_samp1,W)+hb)
  conv_fwd_wmma<<<dim3(8, 248, 16), 256, 0, stream>>>(vsamp1, Apack, hbias,
                                                      nullptr, hprob1, nullptr);
  corr_wmma<<<dim3(31, 16), 256, 0, stream>>>(vhalf, hprob0, accP);    // pos
  corr_wmma<<<dim3(31, 16), 256, 0, stream>>>(vprob1, hprob1, accN);   // neg
  reduce_v_kernel<<<512, 256, 0, stream>>>(v, vsamp1, misc);
  reduce_h_kernel<<<1024, 256, 0, stream>>>(hprob0, hprob1, misc);
  finalize_kernel<<<(15620 + 255) / 256, 256, 0, stream>>>(W, vbias, hbias, accP,
                                                           accN, misc, out);
}
